// FlashAttention3_66486093742327
// MI455X (gfx1250) — matrix operand order repaired and checked by reading, not compiled, not run on hardware
//
#include <hip/hip_runtime.h>

// ---------------------------------------------------------------------------
// Problem constants (from reference): B=4, T=4096, D=1024, H=16, DH=64, BKV=128
// ---------------------------------------------------------------------------
constexpr int Bsz = 4, Tseq = 4096, Dm = 1024, Hn = 16, DHd = 64;

typedef __attribute__((ext_vector_type(16))) __bf16 v16bf;
typedef __attribute__((ext_vector_type(8)))  float  v8f;

union Frag {
  v16bf v;
  uint4 q[2];
};

__device__ __forceinline__ unsigned short f32_to_bf16(float f) {
  union { float f; unsigned u; } cv; cv.f = f;
  unsigned u = cv.u;
  u += 0x7FFFu + ((u >> 16) & 1u);   // round-to-nearest-even
  return (unsigned short)(u >> 16);
}

// --- CDNA5 async global->LDS copy (ASYNCcnt-tracked), via inline asm --------
// LDS byte address for VDST comes from truncating the generic pointer
// (flat LDS aperture: the low 32 bits are the LDS address).
__device__ __forceinline__ void async_copy_b128(void* lds_ptr, const void* gptr) {
  unsigned lds = (unsigned)(uintptr_t)lds_ptr;
  unsigned long long ga = (unsigned long long)(uintptr_t)gptr;
  asm volatile("global_load_async_to_lds_b128 %0, %1, off"
               :: "v"(lds), "v"(ga)
               : "memory");
}
__device__ __forceinline__ void wait_async0() {
  asm volatile("s_wait_asynccnt 0x0" ::: "memory");
}

// ---------------------------------------------------------------------------
// f32 -> bf16 bulk convert (4 elems / thread, grid-stride)
// ---------------------------------------------------------------------------
__global__ __launch_bounds__(256) void cvt_f32_bf16_kernel(
    const float* __restrict__ src, unsigned short* __restrict__ dst, int n4) {
  int i = blockIdx.x * blockDim.x + threadIdx.x;
  int stride = gridDim.x * blockDim.x;
  for (; i < n4; i += stride) {
    float4 f = ((const float4*)src)[i];
    ushort4 h;
    h.x = f32_to_bf16(f.x); h.y = f32_to_bf16(f.y);
    h.z = f32_to_bf16(f.z); h.w = f32_to_bf16(f.w);
    ((ushort4*)dst)[i] = h;
  }
}

// ---------------------------------------------------------------------------
// GEMM: out = A(M,K) @ Bw(N,K)^T + bias, bf16 in, f32 accumulate (WMMA).
// 128x128 macro tile, K-step 32, 8 waves; wave tile 32x64.
// Async double-buffered LDS staging: tile kk+32 streams in while kk computes.
// MODE 0: bf16 out at (B,H,T,DH) | MODE 1: bf16 at (B,H,DH,T) | MODE 2: f32 (M,N)
// ---------------------------------------------------------------------------
template <int MODE>
__global__ __launch_bounds__(256) void gemm_bias_bf16_kernel(
    const unsigned short* __restrict__ A,
    const unsigned short* __restrict__ Bw,
    const float* __restrict__ bias,
    void* __restrict__ outp,
    int M, int N, int K) {
  __shared__ unsigned short As[2][128][40];   // rows 80B -> 16B-aligned
  __shared__ unsigned short Bs[2][128][40];

  const int tid  = threadIdx.x;
  const int wave = tid >> 5, lane = tid & 31;
  const int lm   = lane & 15, hh = lane >> 4;
  const int wy   = wave >> 1, wx = wave & 1;
  const int m0   = blockIdx.y * 128, n0 = blockIdx.x * 128;

  const int lr = tid >> 1;         // 0..127: tile row
  const int lc = (tid & 1) * 16;   // 0 or 16: K-chunk

  auto issue_tile = [&](int buf, int kk) {
    const unsigned short* ga = A + (size_t)(m0 + lr) * K + kk + lc;
    async_copy_b128(&As[buf][lr][lc],     ga);
    async_copy_b128(&As[buf][lr][lc + 8], ga + 8);
    const unsigned short* gb = Bw + (size_t)(n0 + lr) * K + kk + lc;
    async_copy_b128(&Bs[buf][lr][lc],     gb);
    async_copy_b128(&Bs[buf][lr][lc + 8], gb + 8);
  };

  v8f acc[2][4] = {};
  issue_tile(0, 0);

  const int nsteps = K / 32;
  for (int i = 0; i < nsteps; ++i) {
    wait_async0();
    __syncthreads();
    if (i + 1 < nsteps) issue_tile((i + 1) & 1, (i + 1) * 32);
    const int buf = i & 1;

    Frag af[2], bfr[4];
#pragma unroll
    for (int mt = 0; mt < 2; ++mt) {
      int row = wy * 32 + mt * 16 + lm;
      // A 16x32 bf16 layout: lane half h covers K = [8h,8h+8) U [16+8h,16+8h+8)
      af[mt].q[0] = *(const uint4*)&As[buf][row][8 * hh];
      af[mt].q[1] = *(const uint4*)&As[buf][row][16 + 8 * hh];
    }
#pragma unroll
    for (int nt = 0; nt < 4; ++nt) {
      int col = wx * 64 + nt * 16 + lm;
      // B 32x16 bf16 layout: lane half h covers K = [8h,8h+8) U [16+8h,16+8h+8)
      bfr[nt].q[0] = *(const uint4*)&Bs[buf][col][8 * hh];
      bfr[nt].q[1] = *(const uint4*)&Bs[buf][col][16 + 8 * hh];
    }
#pragma unroll
    for (int mt = 0; mt < 2; ++mt)
#pragma unroll
      for (int nt = 0; nt < 4; ++nt)
        acc[mt][nt] = __builtin_amdgcn_wmma_f32_16x16x32_bf16(
            false, af[mt].v, false, bfr[nt].v, (short)0, acc[mt][nt], false, false);
  }

  // Epilogue: C layout -> row = r + 8*half, col = lane%16 per 16x16 tile
#pragma unroll
  for (int mt = 0; mt < 2; ++mt) {
#pragma unroll
    for (int nt = 0; nt < 4; ++nt) {
      int col = n0 + wx * 64 + nt * 16 + lm;
      float bc = bias[col];
#pragma unroll
      for (int r = 0; r < 8; ++r) {
        int row = m0 + wy * 32 + mt * 16 + r + 8 * hh;
        float v = acc[mt][nt][r] + bc;
        if (MODE == 0) {
          int b = row >> 12, t = row & (Tseq - 1);
          int hd = col >> 6, dh = col & 63;
          ((unsigned short*)outp)[(((size_t)b * Hn + hd) * Tseq + t) * DHd + dh] =
              f32_to_bf16(v);
        } else if (MODE == 1) {
          int b = row >> 12, t = row & (Tseq - 1);
          int hd = col >> 6, dh = col & 63;
          ((unsigned short*)outp)[(((size_t)b * Hn + hd) * DHd + dh) * Tseq + t] =
              f32_to_bf16(v);
        } else {
          ((float*)outp)[(size_t)row * N + col] = v;
        }
      }
    }
  }
}

// ---------------------------------------------------------------------------
// Flash attention (faithful to reference quirks: no 1/sqrt(d); O accumulator
// NOT rescaled by exp(M-Mn); final O/L).
// Grid (T/128, H, B), 8 waves; each wave owns 16 query rows.
// K and V tiles for each 128-key block are staged ONCE per workgroup into LDS
// via async copies (double-buffered) -- 8x less L2 traffic than per-wave loads.
// Dynamic LDS: Ks[2][128][72] + Vs[2][64][136] + Ps[8][16][136] = 106496 B.
// ---------------------------------------------------------------------------
__global__ __launch_bounds__(256) void flash_attn_kernel(
    const unsigned short* __restrict__ Qg,
    const unsigned short* __restrict__ Kg,
    const unsigned short* __restrict__ Vt,
    unsigned short* __restrict__ Og) {
  extern __shared__ unsigned short smem[];
  unsigned short* Ks = smem;                    // [2][128][72]
  unsigned short* Vs = Ks + 2 * 128 * 72;       // [2][64][136]
  unsigned short* Ps = Vs + 2 * 64 * 136;       // [8][16][136]

  const int tid  = threadIdx.x;
  const int wave = tid >> 5, lane = tid & 31;
  const int lm   = lane & 15, hh = lane >> 4;
  const int head = blockIdx.y, batch = blockIdx.z;
  const size_t bh = (size_t)batch * Hn + head;
  const unsigned short* Qp = Qg + bh * Tseq * DHd;
  const unsigned short* Kp = Kg + bh * Tseq * DHd;
  const unsigned short* Vp = Vt + bh * DHd * Tseq;
  const int q0 = blockIdx.x * 128 + wave * 16;

  // Issue async copies for kv-block j into buffer buf.
  // K tile: 128 rows x 64 dh (row stride 72), V tile: 64 rows x 128 keys (stride 136).
  auto issue_kv = [&](int buf, int j) {
    const int k0 = j * 128;
#pragma unroll
    for (int i = 0; i < 4; ++i) {
      int c   = tid + i * 256;        // 0..1023
      int row = c >> 3;               // key row
      int col = (c & 7) * 8;          // dh element
      async_copy_b128(Ks + ((size_t)(buf * 128 + row) * 72 + col),
                      Kp + (size_t)(k0 + row) * DHd + col);
    }
#pragma unroll
    for (int i = 0; i < 4; ++i) {
      int c   = tid + i * 256;
      int row = c >> 4;               // dh row
      int col = (c & 15) * 8;         // key element
      async_copy_b128(Vs + ((size_t)(buf * 64 + row) * 136 + col),
                      Vp + (size_t)row * Tseq + k0 + col);
    }
  };

  // Wave's Q A-fragments (2 chunks over dh=64), loaded once from global
  Frag qa[2];
  {
    const unsigned short* qr = Qp + (size_t)(q0 + lm) * DHd;
#pragma unroll
    for (int c = 0; c < 2; ++c) {
      qa[c].q[0] = *(const uint4*)(qr + c * 32 + 8 * hh);
      qa[c].q[1] = *(const uint4*)(qr + c * 32 + 16 + 8 * hh);
    }
  }

  float Mrun[8], Lrun[8];
#pragma unroll
  for (int r = 0; r < 8; ++r) { Mrun[r] = -__builtin_inff(); Lrun[r] = 0.f; }
  v8f o[4] = {};

  issue_kv(0, 0);

  const int nblk = Tseq / 128;
  for (int j = 0; j < nblk; ++j) {
    wait_async0();
    __syncthreads();
    if (j + 1 < nblk) issue_kv((j + 1) & 1, j + 1);
    const int buf = j & 1;
    const unsigned short* Kb = Ks + (size_t)buf * 128 * 72;
    const unsigned short* Vb = Vs + (size_t)buf * 64 * 136;

    // ---- S = Q @ K^T : 8 key tiles x 2 K-chunks = 16 WMMAs ----
    v8f s[8];
#pragma unroll
    for (int nt = 0; nt < 8; ++nt) {
      const unsigned short* kr = Kb + (size_t)(nt * 16 + lm) * 72;
      v8f a = {};
#pragma unroll
      for (int c = 0; c < 2; ++c) {
        Frag kb;
        const uint4* kp = (const uint4*)(kr + c * 32 + 8 * hh);
        kb.q[0] = kp[0];
        kb.q[1] = kp[2];
        a = __builtin_amdgcn_wmma_f32_16x16x32_bf16(
            false, qa[c].v, false, kb.v, (short)0, a, false, false);
      }
      s[nt] = a;
    }

    // ---- running max / exp / running sum (per row, across 16 lanes) ----
    float mn[8], rs[8];
#pragma unroll
    for (int r = 0; r < 8; ++r) {
      float mx = s[0][r];
#pragma unroll
      for (int nt = 1; nt < 8; ++nt) mx = fmaxf(mx, s[nt][r]);
#pragma unroll
      for (int off = 1; off < 16; off <<= 1) mx = fmaxf(mx, __shfl_xor(mx, off, 16));
      mn[r] = fmaxf(Mrun[r], mx);
      rs[r] = 0.f;
    }
#pragma unroll
    for (int nt = 0; nt < 8; ++nt)
#pragma unroll
      for (int r = 0; r < 8; ++r) {
        float e = __expf(s[nt][r] - mn[r]);
        s[nt][r] = e;
        rs[r] += e;
      }
#pragma unroll
    for (int r = 0; r < 8; ++r) {
#pragma unroll
      for (int off = 1; off < 16; off <<= 1) rs[r] += __shfl_xor(rs[r], off, 16);
      Lrun[r] = __expf(Mrun[r] - mn[r]) * Lrun[r] + rs[r];
      Mrun[r] = mn[r];
    }

    // ---- P -> LDS (wave-private tile; same-wave RAW via dscnt waits) ----
    unsigned short* Pw = Ps + (size_t)wave * 16 * 136;
#pragma unroll
    for (int nt = 0; nt < 8; ++nt)
#pragma unroll
      for (int r = 0; r < 8; ++r)
        Pw[(size_t)(r + 8 * hh) * 136 + nt * 16 + lm] = f32_to_bf16(s[nt][r]);

    // ---- O += P @ V : 4 key chunks x 4 dh tiles = 16 WMMAs ----
#pragma unroll
    for (int kc = 0; kc < 4; ++kc) {
      Frag pa;
      const unsigned short* pr = Pw + (size_t)lm * 136;
      pa.q[0] = *(const uint4*)(pr + kc * 32 + 8 * hh);
      pa.q[1] = *(const uint4*)(pr + kc * 32 + 16 + 8 * hh);
#pragma unroll
      for (int nt2 = 0; nt2 < 4; ++nt2) {
        const unsigned short* vr = Vb + (size_t)(nt2 * 16 + lm) * 136 + kc * 32;
        Frag vb;
        const uint4* vp = (const uint4*)(vr + 8 * hh);
        vb.q[0] = vp[0];
        vb.q[1] = vp[2];
        o[nt2] = __builtin_amdgcn_wmma_f32_16x16x32_bf16(
            false, pa.v, false, vb.v, (short)0, o[nt2], false, false);
      }
    }
  }

  // ---- finalize: O / L, store bf16 at (B, T, H*64) ----
#pragma unroll
  for (int r = 0; r < 8; ++r) {
    float inv = 1.0f / Lrun[r];
    int trow = q0 + r + 8 * hh;
    size_t base = ((size_t)batch * Tseq + trow) * Dm + (size_t)head * DHd;
#pragma unroll
    for (int nt2 = 0; nt2 < 4; ++nt2)
      Og[base + nt2 * 16 + lm] = f32_to_bf16(o[nt2][r] * inv);
  }
}

// ---------------------------------------------------------------------------
// Host orchestration.
// Workspace (bf16 elems), total (4*D^2 + 4*M*D)*2 B = 136 MB:
//   wqb wkb wvb wob | xb (reused as attention output) | Q | K | Vt
// ---------------------------------------------------------------------------
extern "C" void kernel_launch(void* const* d_in, const int* in_sizes, int n_in,
                              void* d_out, int out_size, void* d_ws, size_t ws_size,
                              hipStream_t stream) {
  (void)in_sizes; (void)n_in; (void)out_size; (void)ws_size;

  const float* x  = (const float*)d_in[0];
  const float* wq = (const float*)d_in[1];
  const float* bq = (const float*)d_in[2];
  const float* wk = (const float*)d_in[3];
  const float* bk = (const float*)d_in[4];
  const float* wv = (const float*)d_in[5];
  const float* bv = (const float*)d_in[6];
  const float* wo = (const float*)d_in[7];
  const float* bo = (const float*)d_in[8];

  const int M = Bsz * Tseq;            // 16384
  const size_t DD = (size_t)Dm * Dm;   // 1M
  const size_t MD = (size_t)M * Dm;    // 16M

  unsigned short* wqb = (unsigned short*)d_ws;
  unsigned short* wkb = wqb + DD;
  unsigned short* wvb = wkb + DD;
  unsigned short* wvb2= wvb;           // alias readability
  unsigned short* wob = wvb + DD;
  unsigned short* xb  = wob + DD;      // later aliased as attention output
  unsigned short* Qb  = xb + MD;
  unsigned short* Kb  = Qb + MD;
  unsigned short* Vtb = Kb + MD;
  (void)wvb2;

  cvt_f32_bf16_kernel<<<2048, 256, 0, stream>>>(x,  xb,  (int)(MD / 4));
  cvt_f32_bf16_kernel<<<512,  256, 0, stream>>>(wq, wqb, (int)(DD / 4));
  cvt_f32_bf16_kernel<<<512,  256, 0, stream>>>(wk, wkb, (int)(DD / 4));
  cvt_f32_bf16_kernel<<<512,  256, 0, stream>>>(wv, wvb, (int)(DD / 4));
  cvt_f32_bf16_kernel<<<512,  256, 0, stream>>>(wo, wob, (int)(DD / 4));

  dim3 gg(Dm / 128, M / 128);   // (8, 128)
  gemm_bias_bf16_kernel<0><<<gg, 256, 0, stream>>>(xb, wqb, bq, Qb,  M, Dm, Dm);
  gemm_bias_bf16_kernel<0><<<gg, 256, 0, stream>>>(xb, wkb, bk, Kb,  M, Dm, Dm);
  gemm_bias_bf16_kernel<1><<<gg, 256, 0, stream>>>(xb, wvb, bv, Vtb, M, Dm, Dm);

  // Dynamic LDS: Ks[2][128][72] + Vs[2][64][136] + Ps[8][16][136] bf16
  const size_t attn_lds = (size_t)(2 * 128 * 72 + 2 * 64 * 136 + 8 * 16 * 136) * 2;
  flash_attn_kernel<<<dim3(Tseq / 128, Hn, Bsz), 256, attn_lds, stream>>>(Qb, Kb, Vtb, xb);

  gemm_bias_bf16_kernel<2><<<gg, 256, 0, stream>>>(xb, wob, bo, d_out, M, Dm, Dm);
}
